// GCN_1030792151206
// MI455X (gfx1250) — matrix operand order repaired and checked by reading, not compiled, not run on hardware
//
#include <hip/hip_runtime.h>
#include <hip/hip_bf16.h>

typedef __attribute__((ext_vector_type(16))) _Float16 v16h;
typedef __attribute__((ext_vector_type(8)))  float    v8f;
typedef __attribute__((ext_vector_type(8)))  unsigned int v8u;

#define NEG_SLOPE 0.01f
#define BN_EPS 1e-5f

// ---------------------------------------------------------------------------
// Degree / normalization kernels
// ---------------------------------------------------------------------------
__global__ void deg_init_kernel(float* deg, int n) {
    int i = blockIdx.x * blockDim.x + threadIdx.x;
    if (i < n) deg[i] = 1.0f;  // self-loop contributes 1
}

__global__ void deg_accum_kernel(float* deg, const int* __restrict__ dst, int E) {
    int e = blockIdx.x * blockDim.x + threadIdx.x;
    if (e < E) atomicAdd(&deg[dst[e]], 1.0f);
}

__global__ void deg_rsqrt_kernel(float* deg, int n) {
    int i = blockIdx.x * blockDim.x + threadIdx.x;
    if (i < n) deg[i] = rsqrtf(deg[i]);  // deg >= 1 always
}

// ---------------------------------------------------------------------------
// Weight f32 -> f16 conversion with K padding (zeros beyond fi)
// ---------------------------------------------------------------------------
__global__ void wconv_kernel(const float* __restrict__ w, _Float16* o,
                             int fi, int fo, int kpad) {
    int t = blockIdx.x * blockDim.x + threadIdx.x;
    if (t >= kpad * fo) return;
    int k = t / fo, n = t % fo;
    o[t] = (k < fi) ? (_Float16)w[(size_t)k * fo + n] : (_Float16)0.0f;
}

// ---------------------------------------------------------------------------
// CNN encoder: one workgroup (128 threads) per node, all stages in LDS.
// input [1,100,5] -> conv(10,3,3)+leaky -> conv(20,3,2)+leaky -> conv(1,3,2)
// -> tanh -> 94 features, written as f16 into H (stride 128, cols 94.. zeroed)
// ---------------------------------------------------------------------------
__global__ __launch_bounds__(128) void encoder_kernel(
    const float* __restrict__ x,
    const float* __restrict__ w1, const float* __restrict__ b1,
    const float* __restrict__ w2, const float* __restrict__ b2,
    const float* __restrict__ w3, const float* __restrict__ b3,
    _Float16* __restrict__ H) {
    __shared__ float sx[500];          // 100x5 input
    __shared__ float s1[10 * 98 * 3];  // conv1 out
    __shared__ float s2[20 * 96 * 2];  // conv2 out
    __shared__ float sw1[90], sw2[1200], sw3[120];
    __shared__ float sb1[10], sb2[20];

    const int node = blockIdx.x;
    const int tid  = threadIdx.x;
    const float* xin = x + (size_t)node * 500;

    for (int i = tid; i < 500;  i += 128) sx[i]  = xin[i];
    for (int i = tid; i < 90;   i += 128) sw1[i] = w1[i];
    for (int i = tid; i < 1200; i += 128) sw2[i] = w2[i];
    for (int i = tid; i < 120;  i += 128) sw3[i] = w3[i];
    if (tid < 10) sb1[tid] = b1[tid];
    if (tid < 20) sb2[tid] = b2[tid];
    __syncthreads();

    // conv1: out [10,98,3], kernel 3x3 on [100,5]
    for (int idx = tid; idx < 10 * 98 * 3; idx += 128) {
        int co  = idx / 294;
        int rem = idx % 294;
        int oh  = rem / 3, ow = rem % 3;
        float acc = sb1[co];
        const float* wk = sw1 + co * 9;
        #pragma unroll
        for (int kh = 0; kh < 3; ++kh)
            #pragma unroll
            for (int kw = 0; kw < 3; ++kw)
                acc += sx[(oh + kh) * 5 + (ow + kw)] * wk[kh * 3 + kw];
        s1[idx] = (acc > 0.0f) ? acc : NEG_SLOPE * acc;
    }
    __syncthreads();

    // conv2: out [20,96,2], kernel 3x2 over 10 channels of [98,3]
    for (int idx = tid; idx < 20 * 96 * 2; idx += 128) {
        int co  = idx / 192;
        int rem = idx % 192;
        int oh  = rem / 2, ow = rem % 2;
        float acc = sb2[co];
        const float* wk = sw2 + co * 60;
        #pragma unroll
        for (int ci = 0; ci < 10; ++ci) {
            const float* pin = s1 + ci * 294;
            #pragma unroll
            for (int kh = 0; kh < 3; ++kh)
                #pragma unroll
                for (int kw = 0; kw < 2; ++kw)
                    acc += pin[(oh + kh) * 3 + (ow + kw)] * wk[ci * 6 + kh * 2 + kw];
        }
        s2[idx] = (acc > 0.0f) ? acc : NEG_SLOPE * acc;
    }
    __syncthreads();

    // conv3: out [94], kernel 3x2 over 20 channels of [96,2]; then tanh
    _Float16* ho = H + (size_t)node * 128;
    if (tid < 94) {
        float acc = b3[0];
        #pragma unroll
        for (int ci = 0; ci < 20; ++ci) {
            const float* pin = s2 + ci * 192;
            const float* wk  = sw3 + ci * 6;
            #pragma unroll
            for (int kh = 0; kh < 3; ++kh)
                #pragma unroll
                for (int kw = 0; kw < 2; ++kw)
                    acc += pin[(tid + kh) * 2 + kw] * wk[kh * 2 + kw];
        }
        ho[tid] = (_Float16)tanhf(acc);
    } else {
        ho[tid] = (_Float16)0.0f;  // zero pad cols 94..127 (K-padding for GEMM)
    }
}

// ---------------------------------------------------------------------------
// WMMA GEMM: Y[n x fo] = H[n x kpad (stride 128), f16] @ W[kpad x fo, f16]
// One wave per 16x16 output tile, K stepped by 32 via v_wmma_f32_16x16x32_f16.
// A layout per lane (ISA 16-bit A 16x32): row = lane&15; K-halves:
//   lanes 0-15 get K {0..7,16..23}, lanes 16-31 get K {8..15,24..31}.
// B layout: lane holds column N = n0+(lane&15); K order same as A.
// C layout: VGPR r, lanes 0-15 -> M=r, lanes 16-31 -> M=8+r; N = lane&15.
// ---------------------------------------------------------------------------
__global__ __launch_bounds__(32) void gemm_wmma_kernel(
    const _Float16* __restrict__ H, const _Float16* __restrict__ W,
    float* __restrict__ Y, int nrows, int kpad, int fo) {
    const int l  = threadIdx.x;            // 0..31
    const int m0 = blockIdx.x * 16;
    const int n0 = blockIdx.y * 16;

    int arow = m0 + (l & 15);
    if (arow >= nrows) arow = nrows - 1;   // clamp (stores are guarded)
    const _Float16* hrow = H + (size_t)arow * 128;
    const int kb = (l >> 4) << 3;          // 0 or 8

    v8f c = {};
    for (int k0 = 0; k0 < kpad; k0 += 32) {
        // --- A fragment: 8 dword loads of K-pairs ---
        v8u av;
        #pragma unroll
        for (int v = 0; v < 8; ++v) {
            int koff = ((v & 4) ? 16 : 0) + kb + ((v & 3) << 1);
            av[v] = *(const unsigned int*)(hrow + k0 + koff);
        }
        union { v8u u; v16h h; } ca; ca.u = av;

        // --- B fragment: column N = n0+(l&15), K in same order as A ---
        v16h bh;
        #pragma unroll
        for (int s = 0; s < 16; ++s) {
            int kk = ((s & 8) ? 16 : 0) + kb + (s & 7);
            bh[s] = W[(size_t)(k0 + kk) * fo + n0 + (l & 15)];
        }
        union { v8u u; v16h h; } cb; cb.h = bh;

        c = __builtin_amdgcn_wmma_f32_16x16x32_f16(
                false, ca.h, false, cb.h, (short)0, c, false, false);
    }

    const int col   = n0 + (l & 15);
    const int rbase = m0 + ((l >> 4) << 3);
    #pragma unroll
    for (int r = 0; r < 8; ++r) {
        int row = rbase + r;
        if (row < nrows) Y[(size_t)row * fo + col] = c[r];
    }
}

// ---------------------------------------------------------------------------
// Aggregation init: Agg[i][f] = bias[f]  (folds the +b of the GCN layer)
// ---------------------------------------------------------------------------
__global__ void agg_init_kernel(float* Agg, const float* __restrict__ bias,
                                int fo, int total) {
    int t = blockIdx.x * blockDim.x + threadIdx.x;
    if (t < total) Agg[t] = bias[t % fo];
}

// ---------------------------------------------------------------------------
// Edge scatter: Agg[dst] += Y[src] * dinv[src]*dinv[dst]; edges E..E+n-1 are
// self-loops. One thread handles 4 features -> one wave covers 128 features.
// ---------------------------------------------------------------------------
__global__ void scatter_kernel(const float* __restrict__ Y, float* Agg,
                               const int* __restrict__ src,
                               const int* __restrict__ dst,
                               const float* __restrict__ dinv,
                               int E, int n, int fo4) {
    long long t = (long long)blockIdx.x * blockDim.x + threadIdx.x;
    long long total = (long long)(E + n) * fo4;
    if (t >= total) return;
    int e  = (int)(t / fo4);
    int c4 = (int)(t % fo4) * 4;
    int s, d;
    if (e < E) { s = src[e]; d = dst[e]; } else { s = d = e - E; }
    float nm = dinv[s] * dinv[d];
    const float4 y = *(const float4*)(Y + (size_t)s * fo4 * 4 + c4);
    float* o = Agg + (size_t)d * fo4 * 4 + c4;
    atomicAdd(o + 0, y.x * nm);
    atomicAdd(o + 1, y.y * nm);
    atomicAdd(o + 2, y.z * nm);
    atomicAdd(o + 3, y.w * nm);
}

// ---------------------------------------------------------------------------
// Epilogue: BN + tanh -> f16 activation buffer (stride 128) for next GEMM
// ---------------------------------------------------------------------------
__global__ void epilogue_kernel(const float* __restrict__ Agg,
                                const float* __restrict__ g,
                                const float* __restrict__ b,
                                const float* __restrict__ m,
                                const float* __restrict__ v,
                                _Float16* __restrict__ Hn, int fo, int total) {
    int t = blockIdx.x * blockDim.x + threadIdx.x;
    if (t >= total) return;
    int i = t / fo, f = t % fo;
    float x = Agg[t];
    float y = (x - m[f]) * (g[f] * rsqrtf(v[f] + BN_EPS)) + b[f];
    Hn[(size_t)i * 128 + f] = (_Float16)tanhf(y);
}

// ---------------------------------------------------------------------------
// Final layer: BN + softmax over 16 classes
// ---------------------------------------------------------------------------
__global__ void softmax_kernel(const float* __restrict__ Agg,
                               const float* __restrict__ g,
                               const float* __restrict__ b,
                               const float* __restrict__ m,
                               const float* __restrict__ v,
                               float* __restrict__ out, int n) {
    int i = blockIdx.x * blockDim.x + threadIdx.x;
    if (i >= n) return;
    float t[16];
    float mx = -1e30f;
    #pragma unroll
    for (int f = 0; f < 16; ++f) {
        float x = Agg[(size_t)i * 16 + f];
        x = (x - m[f]) * (g[f] * rsqrtf(v[f] + BN_EPS)) + b[f];
        t[f] = x;
        mx = fmaxf(mx, x);
    }
    float s = 0.0f;
    #pragma unroll
    for (int f = 0; f < 16; ++f) { t[f] = __expf(t[f] - mx); s += t[f]; }
    float inv = 1.0f / s;
    #pragma unroll
    for (int f = 0; f < 16; ++f) out[(size_t)i * 16 + f] = t[f] * inv;
}

// ---------------------------------------------------------------------------
// Host launcher
// ---------------------------------------------------------------------------
extern "C" void kernel_launch(void* const* d_in, const int* in_sizes, int n_in,
                              void* d_out, int out_size, void* d_ws, size_t ws_size,
                              hipStream_t stream) {
    const int N = in_sizes[0] / 500;     // 100000
    const int E = in_sizes[1] / 2;       // 1600000

    const float* inputs = (const float*)d_in[0];
    const int*   src    = (const int*)d_in[1];
    const int*   dst    = src + E;
    const float* cw1 = (const float*)d_in[2];
    const float* cb1 = (const float*)d_in[3];
    const float* cw2 = (const float*)d_in[4];
    const float* cb2 = (const float*)d_in[5];
    const float* cw3 = (const float*)d_in[6];
    const float* cb3 = (const float*)d_in[7];

    // per-layer params, setup_inputs() dict order: gw, gb, bn_g, bn_b, bn_m, bn_v
    const float *gw[5], *gb[5], *bg[5], *bb[5], *bm[5], *bv[5];
    for (int i = 0; i < 5; ++i) {
        int base = 8 + 6 * i;
        gw[i] = (const float*)d_in[base + 0];
        gb[i] = (const float*)d_in[base + 1];
        bg[i] = (const float*)d_in[base + 2];
        bb[i] = (const float*)d_in[base + 3];
        bm[i] = (const float*)d_in[base + 4];
        bv[i] = (const float*)d_in[base + 5];
    }

    // layer geometry: fi, kpad (K padded to mult of 32), fo
    const int FI[5]   = {94, 128, 128, 64, 64};
    const int KPAD[5] = {96, 128, 128, 64, 64};
    const int FO[5]   = {128, 128, 64, 64, 16};

    // workspace layout (256B aligned pieces)
    char* p = (char*)d_ws;
    auto take = [&](size_t bytes) {
        char* r = p;
        p += (bytes + 255) & ~(size_t)255;
        return r;
    };
    _Float16* Ha   = (_Float16*)take((size_t)N * 128 * sizeof(_Float16));
    _Float16* Hb   = (_Float16*)take((size_t)N * 128 * sizeof(_Float16));
    float*    Yb   = (float*)take((size_t)N * 128 * sizeof(float));
    float*    Agg  = (float*)take((size_t)N * 128 * sizeof(float));
    float*    dinv = (float*)take((size_t)N * sizeof(float));
    _Float16* Wf[5];
    for (int i = 0; i < 5; ++i)
        Wf[i] = (_Float16*)take((size_t)KPAD[i] * FO[i] * sizeof(_Float16));
    (void)ws_size;

    const int TB = 256;
    auto cdiv = [](long long a, long long b) { return (unsigned)((a + b - 1) / b); };

    // 1. GCN symmetric normalization: dinv = rsqrt(deg + 1)
    deg_init_kernel<<<cdiv(N, TB), TB, 0, stream>>>(dinv, N);
    deg_accum_kernel<<<cdiv(E, TB), TB, 0, stream>>>(dinv, dst, E);
    deg_rsqrt_kernel<<<cdiv(N, TB), TB, 0, stream>>>(dinv, N);

    // 2. weights -> f16 (K-padded)
    for (int i = 0; i < 5; ++i)
        wconv_kernel<<<cdiv((long long)KPAD[i] * FO[i], TB), TB, 0, stream>>>(
            gw[i], Wf[i], FI[i], FO[i], KPAD[i]);

    // 3. CNN encoder -> Ha (f16, stride 128, cols 94..127 zeroed)
    encoder_kernel<<<N, 128, 0, stream>>>(inputs, cw1, cb1, cw2, cb2, cw3, cb3, Ha);

    // 4. five GCN layers
    _Float16* cur = Ha;
    _Float16* nxt = Hb;
    for (int i = 0; i < 5; ++i) {
        const int fo = FO[i], kpad = KPAD[i];

        dim3 ggrid(cdiv(N, 16), fo / 16);
        gemm_wmma_kernel<<<ggrid, 32, 0, stream>>>(cur, Wf[i], Yb, N, kpad, fo);

        agg_init_kernel<<<cdiv((long long)N * fo, TB), TB, 0, stream>>>(
            Agg, gb[i], fo, N * fo);

        long long sthreads = (long long)(E + N) * (fo / 4);
        scatter_kernel<<<cdiv(sthreads, TB), TB, 0, stream>>>(
            Yb, Agg, src, dst, dinv, E, N, fo / 4);

        if (i < 4) {
            epilogue_kernel<<<cdiv((long long)N * fo, TB), TB, 0, stream>>>(
                Agg, bg[i], bb[i], bm[i], bv[i], nxt, fo, N * fo);
            _Float16* t = cur; cur = nxt; nxt = t;
        } else {
            softmax_kernel<<<cdiv(N, TB), TB, 0, stream>>>(
                Agg, bg[i], bb[i], bm[i], bv[i], (float*)d_out, N);
        }
    }
}
